// SlotAttentionModel_14757507629951
// MI455X (gfx1250) — matrix operand order repaired and checked by reading, not compiled, not run on hardware
//
#include <hip/hip_runtime.h>
#include <hip/hip_bf16.h>

typedef _Float16 half_t;
typedef __attribute__((ext_vector_type(16))) _Float16 v16h;
typedef __attribute__((ext_vector_type(8)))  float    v8f;

// ---------------------------------------------------------------------------
// Weight prep: convert f32 OIHW (or [N,K] fc) weights into WMMA B-fragment
// swizzled f16 layout.  Fragment (tap t, kchunk c, ntile n): 32 lanes x 16
// halves contiguous; lane l element e = W[k][col],
//   k   = c*32 + ((l&16)?16:0) + e
//   col = n*16 + (l&15)
// flipT=1 => transposed-conv weights: src[ci][co][KH-1-ky][KW-1-kx]
// ---------------------------------------------------------------------------
__global__ void prep_w(const float* __restrict__ src, half_t* __restrict__ dst,
                       int Cin, int Cout, int KH, int KW, int kchunks,
                       int flipT, int total)
{
    int idx = blockIdx.x * 256 + threadIdx.x;
    if (idx >= total) return;
    int e    = idx & 15;
    int l    = (idx >> 4) & 31;
    int frag = idx >> 9;
    int n  = frag & 3;
    int tc = frag >> 2;
    int c  = tc % kchunks;
    int t  = tc / kchunks;
    int ky = t / KW, kx = t - ky * KW;
    int k  = c * 32 + ((l & 16) ? 16 : 0) + e;
    int co = n * 16 + (l & 15);
    float v = 0.f;
    if (k < Cin && co < Cout) {
        if (flipT)
            v = src[(size_t)k * Cout * KH * KW + (size_t)co * KH * KW +
                    (KH - 1 - ky) * KW + (KW - 1 - kx)];
        else
            v = src[(size_t)co * Cin * KH * KW + (size_t)k * KH * KW +
                    ky * KW + kx];
    }
    dst[idx] = (half_t)v;
}

// ---------------------------------------------------------------------------
// Templated WMMA conv / GEMM over pre-padded NHWC f16 activations.
// Fully unrolled taps x kchunks; each wave computes TWO 16-pixel x 16-co
// tiles (acc0/acc1) reusing each B fragment for 2 WMMAs.
// grid = (pixTiles32, imgs), block = 128 (4 waves = 4 co-tiles).
// ---------------------------------------------------------------------------
template <int KH, int KW, int KCH>
__global__ __launch_bounds__(128)
void conv_wmma_t(const half_t* __restrict__ in, const half_t* __restrict__ wfrag,
                 const float* __restrict__ bias, void* __restrict__ outv,
                 int Hp, int Wp, int Wout, int wShift,
                 int HpO, int WpO, int o0, int os,
                 int act, int outF32)
{
    constexpr int CinP = KCH * 32;
    int lane  = threadIdx.x & 31;
    int ntile = threadIdx.x >> 5;
    int img   = blockIdx.y;
    int tile  = blockIdx.x;                 // 32-pixel tile

    int row = lane & 15;
    int kb  = (lane & 16) ? 16 : 0;
    int p0  = tile * 32 + row;
    int p1  = p0 + 16;

    const half_t* inImg = in + (size_t)img * Hp * Wp * CinP;
    const half_t* base0 = inImg + ((size_t)(p0 >> wShift) * Wp + (p0 & (Wout - 1))) * CinP + kb;
    const half_t* base1 = inImg + ((size_t)(p1 >> wShift) * Wp + (p1 & (Wout - 1))) * CinP + kb;

    v8f acc0 = {}, acc1 = {};
    const half_t* wlane = wfrag + lane * 16;

#pragma unroll
    for (int t = 0; t < KH * KW; ++t) {
        const int ky = t / KW, kx = t - ky * KW;
        const size_t tofs = ((size_t)ky * Wp + kx) * CinP;
#pragma unroll
        for (int c = 0; c < KCH; ++c) {
            union { v16h v; uint4 q[2]; } a0, a1, b;
            const uint4* bp = (const uint4*)(wlane + ((size_t)(t * KCH + c) * 4 + ntile) * 512);
            b.q[0] = bp[0];
            b.q[1] = bp[1];
            const uint4* a0p = (const uint4*)(base0 + tofs + c * 32);
            a0.q[0] = a0p[0];               // K kb..kb+7
            a0.q[1] = a0p[1];               // K kb+8..kb+15
            const uint4* a1p = (const uint4*)(base1 + tofs + c * 32);
            a1.q[0] = a1p[0];
            a1.q[1] = a1p[1];
            acc0 = __builtin_amdgcn_wmma_f32_16x16x32_f16(
                       false, a0.v, false, b.v, (short)0, acc0, false, false);
            acc1 = __builtin_amdgcn_wmma_f32_16x16x32_f16(
                       false, a1.v, false, b.v, (short)0, acc1, false, false);
        }
    }

    int   co = ntile * 16 + (lane & 15);
    float bv = bias[co];
    int mofs = (lane & 16) ? 8 : 0;
#pragma unroll
    for (int half = 0; half < 2; ++half) {
        v8f acc = half ? acc1 : acc0;
#pragma unroll
        for (int r = 0; r < 8; ++r) {
            int p2 = tile * 32 + half * 16 + r + mofs;
            int y2 = p2 >> wShift;
            int x2 = p2 & (Wout - 1);
            float val = acc[r] + bv;
            if (act == 1) val = val > 0.f ? val : 0.01f * val;
            size_t o = ((size_t)img * HpO * WpO +
                        (size_t)(o0 + y2 * os) * WpO + (o0 + x2 * os)) * 64 + co;
            if (outF32) ((float*)outv)[o] = val;
            else        ((half_t*)outv)[o] = (half_t)val;
        }
    }
}

// --------------------------- small kernels ---------------------------------
__global__ void stage_input(const float* __restrict__ x, half_t* __restrict__ dst)
{
    int idx = blockIdx.x * 256 + threadIdx.x;    // 16*16384
    if (idx >= 16 * 16384) return;
    int b = idx >> 14, yx = idx & 16383;
    int y = yx >> 7, xx = yx & 127;
    size_t o = ((size_t)b * 132 * 132 + (size_t)(y + 2) * 132 + (xx + 2)) * 32;
    for (int c = 0; c < 3; ++c)
        dst[o + c] = (half_t)x[((size_t)b * 3 + c) * 16384 + yx];
}

__global__ __launch_bounds__(64)
void add_pos(half_t* __restrict__ feat, const float* __restrict__ pw,
             const float* __restrict__ pb)
{
    int pix = blockIdx.x, d = threadIdx.x;
    int yx = pix & 16383;
    int y = yx >> 7, x = yx & 127;
    float gy = y * (1.f / 127.f), gx = x * (1.f / 127.f);
    float pos = pb[d] + pw[d*4+0]*gy + pw[d*4+1]*gx +
                pw[d*4+2]*(1.f-gy) + pw[d*4+3]*(1.f-gx);
    size_t o = (size_t)pix * 64 + d;
    feat[o] = (half_t)((float)feat[o] + pos);
}

__global__ __launch_bounds__(256)
void ln_rows(const float* __restrict__ in, half_t* __restrict__ out,
             const float* __restrict__ g, const float* __restrict__ b, int M)
{
    int lane = threadIdx.x & 31;
    int row  = blockIdx.x * 8 + (threadIdx.x >> 5);
    if (row >= M) return;
    const float* r = in + (size_t)row * 64;
    float x0 = r[lane*2], x1 = r[lane*2+1];
    float s = x0 + x1, sq = x0*x0 + x1*x1;
    for (int o = 16; o > 0; o >>= 1) { s += __shfl_xor(s, o, 32); sq += __shfl_xor(sq, o, 32); }
    float mean = s * (1.f/64.f);
    float inv  = rsqrtf(sq * (1.f/64.f) - mean*mean + 1e-5f);
    int d0 = lane * 2;
    out[(size_t)row*64 + d0]     = (half_t)((x0-mean)*inv*g[d0]   + b[d0]);
    out[(size_t)row*64 + d0 + 1] = (half_t)((x1-mean)*inv*g[d0+1] + b[d0+1]);
}

__global__ void init_slots(const float* __restrict__ mu, const float* __restrict__ ls,
                           const float* __restrict__ noise, float* __restrict__ slots)
{
    int i = blockIdx.x * 256 + threadIdx.x;
    if (i >= 7168) return;
    int d = i & 63;
    slots[i] = mu[d] + expf(ls[d]) * noise[i];
}

__global__ __launch_bounds__(64)
void slot_q(const float* __restrict__ slots, const float* __restrict__ g,
            const float* __restrict__ b, const float* __restrict__ qw,
            float* __restrict__ q)
{
    __shared__ float sh[64];
    __shared__ float red[2];
    int t = threadIdx.x, r = blockIdx.x;
    float x = slots[r*64 + t];
    sh[t] = x; __syncthreads();
    if (t == 0) {
        float s = 0, sq = 0;
        for (int i = 0; i < 64; ++i) { s += sh[i]; sq += sh[i]*sh[i]; }
        float m = s * (1.f/64.f);
        red[0] = m; red[1] = rsqrtf(sq * (1.f/64.f) - m*m + 1e-5f);
    }
    __syncthreads();
    float xn = (x - red[0]) * red[1] * g[t] + b[t];
    __syncthreads();
    sh[t] = xn; __syncthreads();
    float acc = 0.f;
    for (int k = 0; k < 64; ++k) acc += sh[k] * qw[t*64 + k];
    q[r*64 + t] = acc * 0.125f;   // D^-0.5
}

__global__ __launch_bounds__(256)
void attn_p1(const half_t* __restrict__ k, const float* __restrict__ q,
             float* __restrict__ colsum)
{
    __shared__ float qs[7][64];
    __shared__ float bs[7];
    int b = blockIdx.x >> 6;
    int n = ((blockIdx.x & 63) << 8) + threadIdx.x;
    if (threadIdx.x < 7) bs[threadIdx.x] = 0.f;
    for (int i = threadIdx.x; i < 448; i += 256) qs[i>>6][i&63] = q[b*448 + i];
    __syncthreads();
    const half_t* kr = k + ((size_t)b * 16384 + n) * 64;
    float kv[64];
    for (int d = 0; d < 64; ++d) kv[d] = (float)kr[d];
    float lg[7], mx = -1e30f;
    for (int s = 0; s < 7; ++s) {
        float a = 0.f;
        for (int d = 0; d < 64; ++d) a += kv[d] * qs[s][d];
        lg[s] = a; mx = fmaxf(mx, a);
    }
    float sum = 0.f;
    for (int s = 0; s < 7; ++s) { lg[s] = expf(lg[s] - mx); sum += lg[s]; }
    float inv = 1.f / sum;
    for (int s = 0; s < 7; ++s) atomicAdd(&bs[s], lg[s]*inv + 1e-8f);
    __syncthreads();
    if (threadIdx.x < 7) atomicAdd(&colsum[b*7 + threadIdx.x], bs[threadIdx.x]);
}

__global__ __launch_bounds__(256)
void attn_p2(const half_t* __restrict__ k, const half_t* __restrict__ v,
             const float* __restrict__ q, const float* __restrict__ colsum,
             float* __restrict__ upd)
{
    __shared__ float qs[7][64];
    __shared__ float acc[7][64];
    __shared__ float cs[7];
    int b = blockIdx.x >> 6;
    int n = ((blockIdx.x & 63) << 8) + threadIdx.x;
    for (int i = threadIdx.x; i < 448; i += 256) {
        qs[i>>6][i&63] = q[b*448 + i];
        acc[i>>6][i&63] = 0.f;
    }
    if (threadIdx.x < 7) cs[threadIdx.x] = colsum[b*7 + threadIdx.x];
    __syncthreads();
    const half_t* kr = k + ((size_t)b * 16384 + n) * 64;
    const half_t* vr = v + ((size_t)b * 16384 + n) * 64;
    float kv[64];
    for (int d = 0; d < 64; ++d) kv[d] = (float)kr[d];
    float lg[7], mx = -1e30f;
    for (int s = 0; s < 7; ++s) {
        float a = 0.f;
        for (int d = 0; d < 64; ++d) a += kv[d] * qs[s][d];
        lg[s] = a; mx = fmaxf(mx, a);
    }
    float sum = 0.f;
    for (int s = 0; s < 7; ++s) { lg[s] = expf(lg[s] - mx); sum += lg[s]; }
    float inv = 1.f / sum;
    float w[7];
    for (int s = 0; s < 7; ++s) w[s] = (lg[s]*inv + 1e-8f) / cs[s];
    for (int d = 0; d < 64; ++d) {
        float vv = (float)vr[d];
        for (int s = 0; s < 7; ++s) atomicAdd(&acc[s][d], w[s] * vv);
    }
    __syncthreads();
    for (int i = threadIdx.x; i < 448; i += 256)
        atomicAdd(&upd[b*448 + i], acc[i>>6][i&63]);
}

__global__ __launch_bounds__(192)
void slot_update(float* __restrict__ slots, const float* __restrict__ upd,
                 const float* __restrict__ wih, const float* __restrict__ whh,
                 const float* __restrict__ bih, const float* __restrict__ bhh,
                 const float* __restrict__ lng, const float* __restrict__ lnb,
                 const float* __restrict__ w1, const float* __restrict__ b1,
                 const float* __restrict__ w2, const float* __restrict__ b2)
{
    __shared__ float xr[64], hr[64], gi[192], gh[192], hn[64], m[64], m1[128];
    __shared__ float red[2];
    int t = threadIdx.x, rix = blockIdx.x;
    if (t < 64) { xr[t] = upd[rix*64 + t]; hr[t] = slots[rix*64 + t]; }
    __syncthreads();
    { float a = bih[t], c = bhh[t];
      for (int k = 0; k < 64; ++k) { a += xr[k]*wih[t*64+k]; c += hr[k]*whh[t*64+k]; }
      gi[t] = a; gh[t] = c; }
    __syncthreads();
    if (t < 64) {
        float r = 1.f/(1.f+expf(-(gi[t]      + gh[t])));
        float z = 1.f/(1.f+expf(-(gi[64+t]   + gh[64+t])));
        float nn = tanhf(gi[128+t] + r*gh[128+t]);
        hn[t] = (1.f - z)*nn + z*hr[t];
    }
    __syncthreads();
    if (t == 0) {
        float s = 0, sq = 0;
        for (int i = 0; i < 64; ++i) { s += hn[i]; sq += hn[i]*hn[i]; }
        float mm = s*(1.f/64.f);
        red[0] = mm; red[1] = rsqrtf(sq*(1.f/64.f) - mm*mm + 1e-5f);
    }
    __syncthreads();
    if (t < 64) m[t] = (hn[t]-red[0])*red[1]*lng[t] + lnb[t];
    __syncthreads();
    if (t < 128) {
        float a = b1[t];
        for (int k = 0; k < 64; ++k) a += m[k]*w1[t*64+k];
        m1[t] = fmaxf(a, 0.f);
    }
    __syncthreads();
    if (t < 64) {
        float a = b2[t];
        for (int k = 0; k < 128; ++k) a += m1[k]*w2[t*128+k];
        slots[rix*64 + t] = hn[t] + a;
    }
}

__global__ __launch_bounds__(64)
void dec_bc(const float* __restrict__ slots, const float* __restrict__ pw,
            const float* __restrict__ pb, half_t* __restrict__ dst)
{
    int bs = blockIdx.x >> 6;
    int ij = blockIdx.x & 63;
    int i = ij >> 3, j = ij & 7;
    int d = threadIdx.x;
    float gy = i * (1.f/7.f), gx = j * (1.f/7.f);
    float pos = pb[d] + pw[d*4+0]*gy + pw[d*4+1]*gx +
                pw[d*4+2]*(1.f-gy) + pw[d*4+3]*(1.f-gx);
    size_t o = ((size_t)bs*20*20 + (size_t)(2 + 2*i)*20 + (2 + 2*j))*64 + d;
    dst[o] = (half_t)(slots[bs*64 + d] + pos);
}

// dec5: 3x3 convT (stride1, pad1) 64->4, full f32, flipped/transposed weights.
__global__ __launch_bounds__(256)
void dec5_conv(const half_t* __restrict__ in, const float* __restrict__ w,
               const float* __restrict__ bias, float* __restrict__ out)
{
    int idx = blockIdx.x * 256 + threadIdx.x;   // 112*16384
    int bs = idx >> 14, yx = idx & 16383;
    int y = yx >> 7, x = yx & 127;
    const half_t* base = in + ((size_t)bs*130*130 + (size_t)y*130 + x)*64;
    float acc0 = bias[0], acc1 = bias[1], acc2 = bias[2], acc3 = bias[3];
    for (int ky = 0; ky < 3; ++ky)
      for (int kx = 0; kx < 3; ++kx) {
        const half_t* pp = base + ((size_t)ky*130 + kx)*64;
        int wsp = (2-ky)*3 + (2-kx);
        for (int ci = 0; ci < 64; ++ci) {
            float a = (float)pp[ci];
            const float* wc = w + (size_t)ci*36 + wsp;   // src[ci][o][2-ky][2-kx]
            acc0 += a*wc[0]; acc1 += a*wc[9]; acc2 += a*wc[18]; acc3 += a*wc[27];
        }
      }
    float* op = out + (size_t)idx*4;
    op[0]=acc0; op[1]=acc1; op[2]=acc2; op[3]=acc3;
}

__global__ __launch_bounds__(256)
void combine(const float* __restrict__ logits, float* __restrict__ out)
{
    int idx = blockIdx.x * 256 + threadIdx.x;   // b*16384 + yx
    int b = idx >> 14, yx = idx & 16383;
    float a3[7], r0[7], r1[7], r2[7];
    float mx = -1e30f;
    for (int s = 0; s < 7; ++s) {
        const float* l = logits + (((size_t)(b*7 + s))*16384 + yx)*4;
        r0[s]=l[0]; r1[s]=l[1]; r2[s]=l[2]; a3[s]=l[3];
        mx = fmaxf(mx, a3[s]);
    }
    float sum = 0.f;
    for (int s = 0; s < 7; ++s) { a3[s] = expf(a3[s]-mx); sum += a3[s]; }
    float inv = 1.f / sum;
    float rc0=0, rc1=0, rc2=0;
    float* recons = out + 786432;
    float* masks  = out + 786432 + 5505024;
    for (int s = 0; s < 7; ++s) {
        float mk = a3[s]*inv;
        rc0 += mk*r0[s]; rc1 += mk*r1[s]; rc2 += mk*r2[s];
        size_t rb = ((size_t)(b*7+s))*3*16384 + yx;
        recons[rb] = r0[s]; recons[rb+16384] = r1[s]; recons[rb+32768] = r2[s];
        masks[(size_t)(b*7+s)*16384 + yx] = mk;
    }
    out[(size_t)b*3*16384 + yx]         = rc0;
    out[(size_t)b*3*16384 + 16384 + yx] = rc1;
    out[(size_t)b*3*16384 + 32768 + yx] = rc2;
}

// ---------------------------------------------------------------------------
extern "C" void kernel_launch(void* const* d_in, const int* in_sizes, int n_in,
                              void* d_out, int out_size, void* d_ws, size_t ws_size,
                              hipStream_t stream)
{
    (void)in_sizes; (void)n_in; (void)out_size; (void)ws_size;
    const float* X      = (const float*)d_in[0];
    const float* NOISE  = (const float*)d_in[1];
    auto P = [&](int i) { return (const float*)d_in[i]; };

    // ---- workspace layout ----
    char* ws = (char*)d_ws;
    size_t off = 0;
    auto alloc = [&](size_t bytes) { size_t o = off; off += (bytes + 255) & ~(size_t)255; return o; };

    size_t WF_ENC0 = alloc((size_t)25*1*4*512*2);
    size_t WF_ENC1 = alloc((size_t)25*2*4*512*2);
    size_t WF_ENC2 = alloc((size_t)25*2*4*512*2);
    size_t WF_ENC3 = alloc((size_t)25*2*4*512*2);
    size_t WF_FC1  = alloc((size_t)2*4*512*2);
    size_t WF_FC2  = alloc((size_t)2*4*512*2);
    size_t WF_K    = alloc((size_t)2*4*512*2);
    size_t WF_V    = alloc((size_t)2*4*512*2);
    size_t WF_DEC[5];
    for (int i = 0; i < 5; ++i) WF_DEC[i] = alloc((size_t)25*2*4*512*2);

    const size_t BIG = (size_t)112*132*132*64*2;       // ~250 MB
    size_t OFF_A = alloc(BIG);
    size_t OFF_B = alloc(BIG);
    size_t OFF_C = alloc((size_t)262144*64*4);         // 67 MB (feat2 f32 / logits)
    size_t OFF_SLOTS = alloc(7168*4);
    size_t OFF_Q     = alloc(7168*4);
    size_t OFF_CS    = alloc(112*4);
    size_t OFF_UPD   = alloc(7168*4);

    half_t* A  = (half_t*)(ws + OFF_A);
    half_t* Bf = (half_t*)(ws + OFF_B);
    float*  C  = (float*)(ws + OFF_C);
    float*  SLOTS = (float*)(ws + OFF_SLOTS);
    float*  Q     = (float*)(ws + OFF_Q);
    float*  CS    = (float*)(ws + OFF_CS);
    float*  UPD   = (float*)(ws + OFF_UPD);

    // ---- weight prep (every call; deterministic) ----
    auto prep = [&](int srcIdx, size_t dstOff, int Cin, int Cout, int KH, int KW,
                    int kchunks, int flipT) {
        int total = KH*KW*kchunks*4*512;
        prep_w<<<(total+255)/256, 256, 0, stream>>>(P(srcIdx), (half_t*)(ws + dstOff),
                                                    Cin, Cout, KH, KW, kchunks, flipT, total);
    };
    prep(2,  WF_ENC0, 3, 64, 5, 5, 1, 0);
    prep(4,  WF_ENC1, 64, 64, 5, 5, 2, 0);
    prep(6,  WF_ENC2, 64, 64, 5, 5, 2, 0);
    prep(8,  WF_ENC3, 64, 64, 5, 5, 2, 0);
    prep(12, WF_FC1,  64, 64, 1, 1, 2, 0);
    prep(14, WF_FC2,  64, 64, 1, 1, 2, 0);
    prep(23, WF_K,    64, 64, 1, 1, 2, 0);
    prep(25, WF_V,    64, 64, 1, 1, 2, 0);
    prep(37, WF_DEC[0], 64, 64, 5, 5, 2, 1);
    prep(39, WF_DEC[1], 64, 64, 5, 5, 2, 1);
    prep(41, WF_DEC[2], 64, 64, 5, 5, 2, 1);
    prep(43, WF_DEC[3], 64, 64, 5, 5, 2, 1);
    prep(45, WF_DEC[4], 64, 64, 5, 5, 2, 1);

    // conv dispatch: KH,KW,KCH compile-time; grid = (pixTiles32, imgs)
    auto conv = [&](int imgs, const half_t* in, size_t wfOff, const float* bias, void* out,
                    int Hp, int Wp, int CinP, int KH, int Wout, int wShift,
                    int HpO, int WpO, int o0, int os, int act, int outF32) {
        dim3 g((Wout * Wout) / 32, imgs);
        const half_t* wf = (const half_t*)(ws + wfOff);
        if (KH == 5 && CinP == 64)
            conv_wmma_t<5,5,2><<<g, 128, 0, stream>>>(in, wf, bias, out,
                Hp, Wp, Wout, wShift, HpO, WpO, o0, os, act, outF32);
        else if (KH == 5)
            conv_wmma_t<5,5,1><<<g, 128, 0, stream>>>(in, wf, bias, out,
                Hp, Wp, Wout, wShift, HpO, WpO, o0, os, act, outF32);
        else
            conv_wmma_t<1,1,2><<<g, 128, 0, stream>>>(in, wf, bias, out,
                Hp, Wp, Wout, wShift, HpO, WpO, o0, os, act, outF32);
    };

    // ================= encoder =================
    hipMemsetAsync(A, 0, (size_t)16*132*132*32*2, stream);
    stage_input<<<1024, 256, 0, stream>>>(X, A);

    hipMemsetAsync(Bf, 0, (size_t)16*132*132*64*2, stream);
    conv(16, A, WF_ENC0, P(3), Bf, 132,132,32, 5, 128,7, 132,132, 2,1, 1,0);
    hipMemsetAsync(A, 0, (size_t)16*132*132*64*2, stream);
    conv(16, Bf, WF_ENC1, P(5), A, 132,132,64, 5, 128,7, 132,132, 2,1, 1,0);
    hipMemsetAsync(Bf, 0, (size_t)16*132*132*64*2, stream);
    conv(16, A, WF_ENC2, P(7), Bf, 132,132,64, 5, 128,7, 132,132, 2,1, 1,0);
    conv(16, Bf, WF_ENC3, P(9), A, 132,132,64, 5, 128,7, 128,128, 0,1, 1,0); // feat0 in A

    add_pos<<<16*16384, 64, 0, stream>>>(A, P(10), P(11));

    conv(16, A,  WF_FC1, P(13), Bf, 128,128,64, 1, 128,7, 128,128, 0,1, 1,0); // feat1
    conv(16, Bf, WF_FC2, P(15), C,  128,128,64, 1, 128,7, 128,128, 0,1, 0,1); // feat2 f32
    ln_rows<<<262144/8, 256, 0, stream>>>(C, Bf, P(16), P(17), 262144);       // inp in B
    conv(16, Bf, WF_K, P(24), A, 128,128,64, 1, 128,7, 128,128, 0,1, 0,0);    // k in A
    conv(16, Bf, WF_V, P(26), (half_t*)C, 128,128,64, 1, 128,7, 128,128, 0,1, 0,0); // v in C

    // ================= slot attention =================
    init_slots<<<(7168+255)/256, 256, 0, stream>>>(P(35), P(36), NOISE, SLOTS);
    for (int it = 0; it < 4; ++it) {
        slot_q<<<112, 64, 0, stream>>>(SLOTS, P(18), P(19), P(22), Q);
        hipMemsetAsync(CS, 0, 112*4, stream);
        hipMemsetAsync(UPD, 0, 7168*4, stream);
        attn_p1<<<1024, 256, 0, stream>>>(A, Q, CS);
        attn_p2<<<1024, 256, 0, stream>>>(A, (const half_t*)C, Q, CS, UPD);
        slot_update<<<112, 192, 0, stream>>>(SLOTS, UPD,
            P(27), P(28), P(29), P(30), P(20), P(21), P(31), P(32), P(33), P(34));
    }
    hipMemcpyAsync((float*)d_out + 8126464, SLOTS, 7168*4,
                   hipMemcpyDeviceToDevice, stream);

    // ================= decoder =================
    hipMemsetAsync(Bf, 0, (size_t)112*20*20*64*2, stream);
    dec_bc<<<112*64, 64, 0, stream>>>(SLOTS, P(49), P(50), Bf);

    hipMemsetAsync(A, 0, (size_t)112*36*36*64*2, stream);
    conv(112, Bf, WF_DEC[0], P(38), A, 20,20,64, 5, 16,4, 36,36, 2,2, 1,0);
    hipMemsetAsync(Bf, 0, (size_t)112*68*68*64*2, stream);
    conv(112, A, WF_DEC[1], P(40), Bf, 36,36,64, 5, 32,5, 68,68, 2,2, 1,0);
    hipMemsetAsync(A, 0, BIG, stream);
    conv(112, Bf, WF_DEC[2], P(42), A, 68,68,64, 5, 64,6, 132,132, 2,2, 1,0);
    hipMemsetAsync(Bf, 0, BIG, stream);
    conv(112, A, WF_DEC[3], P(44), Bf, 132,132,64, 5, 128,7, 132,132, 2,1, 1,0);
    hipMemsetAsync(A, 0, (size_t)112*130*130*64*2, stream);
    conv(112, Bf, WF_DEC[4], P(46), A, 132,132,64, 5, 128,7, 130,130, 1,1, 1,0);

    dec5_conv<<<112*16384/256, 256, 0, stream>>>(A, P(47), P(48), C);
    combine<<<16*16384/256, 256, 0, stream>>>(C, (float*)d_out);
}
